// ProteinFeaturesLigand_41188736368932
// MI455X (gfx1250) — compile-verified, matrix operand order checked by reading, not run on hardware
//
#include <hip/hip_runtime.h>
#include <hip/hip_bf16.h>

typedef __attribute__((ext_vector_type(16))) _Float16 v16h;
typedef __attribute__((ext_vector_type(8)))  _Float16 v8h;
typedef __attribute__((ext_vector_type(2)))  _Float16 v2h;
typedef __attribute__((ext_vector_type(8)))  float    v8f;

#define BB 2
#define LL 2048
#define KK 32
#define HIDN 128
#define NRBF 16
#define NPOSE 16
#define FEAT 416          // NPOSE + 25*NRBF
#define FPAD 424          // LDS row stride in halfs: 848B = 16B-aligned, conflict-free b128
#define MAXREL 32

// ---------------------------------------------------------------------------
// Kernel 1: build atoms5 = [N, CA, C, O, CB] per residue (15 floats)
// ---------------------------------------------------------------------------
__global__ void geom_kernel(const float* __restrict__ X,
                            float* __restrict__ atoms5, int n) {
    int g = blockIdx.x * blockDim.x + threadIdx.x;
    if (g >= n) return;
    const float* x = X + (size_t)g * 12;
    float Nx = x[0], Ny = x[1], Nz = x[2];
    float Ax = x[3], Ay = x[4], Az = x[5];
    float Cx = x[6], Cy = x[7], Cz = x[8];
    float bx = Ax - Nx, by = Ay - Ny, bz = Az - Nz;
    float cx = Cx - Ax, cy = Cy - Ay, cz = Cz - Az;
    float ax = by * cz - bz * cy;
    float ay = bz * cx - bx * cz;
    float az = bx * cy - by * cx;
    const float cA = -0.58273431f, cB = 0.56802827f, cC = -0.54067466f;
    float* o = atoms5 + (size_t)g * 15;
#pragma unroll
    for (int i = 0; i < 12; ++i) o[i] = x[i];
    o[12] = cA * ax + cB * bx + cC * cx + Ax;
    o[13] = cA * ay + cB * by + cC * cy + Ay;
    o[14] = cA * az + cB * bz + cC * cz + Az;
}

// ---------------------------------------------------------------------------
// Kernel 2: W_edge f32 -> f16 (row-major [128][416], 16B-aligned rows)
// ---------------------------------------------------------------------------
__global__ void cvt_kernel(const float* __restrict__ W,
                           _Float16* __restrict__ W16, int n) {
    int i = blockIdx.x * blockDim.x + threadIdx.x;
    if (i < n) W16[i] = (_Float16)W[i];
}

// ---------------------------------------------------------------------------
// Kernel 3: top-32 neighbors, one wave per query residue
// ---------------------------------------------------------------------------
__global__ void __launch_bounds__(128)
topk_kernel(const float* __restrict__ atoms5,
            const float* __restrict__ mask, int* __restrict__ eidx) {
    __shared__ float dist[4][LL];
    int tid = threadIdx.x, lane = tid & 31, wv = tid >> 5;
    int q = blockIdx.x * 4 + wv;               // global residue (b*L + i)
    int b = q / LL;
    const float* qa = atoms5 + (size_t)q * 15 + 3;   // CA of query
    float qx = qa[0], qy = qa[1], qz = qa[2];
    const float* base = atoms5 + (size_t)b * LL * 15;
    const float* mb = mask + (size_t)b * LL;
    for (int j = lane; j < LL; j += 32) {
        const float* p = base + (size_t)j * 15 + 3;
        float dx = qx - p[0], dy = qy - p[1], dz = qz - p[2];
        dist[wv][j] = dx * dx + dy * dy + dz * dz + (1.0f - mb[j]) * 1e6f;
    }
    __syncthreads();
    for (int it = 0; it < KK; ++it) {
        float best = 3.0e38f;
        int bj = 0x7fffffff;
        for (int j = lane; j < LL; j += 32) {
            float d = dist[wv][j];
            if (d < best) { best = d; bj = j; }   // ascending j => lowest-idx tie
        }
#pragma unroll
        for (int m = 16; m >= 1; m >>= 1) {
            float ob = __shfl_xor(best, m, 32);
            int   oj = __shfl_xor(bj, m, 32);
            if (ob < best || (ob == best && oj < bj)) { best = ob; bj = oj; }
        }
        if (lane == 0) {
            eidx[(size_t)q * KK + it] = bj;
            dist[wv][bj] = 3.0e38f;
        }
        __syncthreads();
    }
}

// ---------------------------------------------------------------------------
// Kernel 4: fused edge features -> f16 LDS tile -> WMMA GEMM -> LayerNorm
// One block (4 waves / 128 threads) = 2 residues = 64 edge rows x 128 outputs
// ---------------------------------------------------------------------------
__global__ void __launch_bounds__(128)
edge_kernel(const float* __restrict__ atoms5,
            const _Float16* __restrict__ wedge,
            const int* __restrict__ Ridx,
            const int* __restrict__ chain,
            const float* __restrict__ Wpos,
            const float* __restrict__ bpos,
            const float* __restrict__ gam,
            const float* __restrict__ bet,
            const int* __restrict__ eidx,
            float* __restrict__ Eout) {
    __shared__ _Float16 eh[64 * FPAD];
    const int tid = threadIdx.x;
    const int blk = blockIdx.x;

    // ---------------- Phase 1: build 64 x 416 f16 feature tile -------------
    {
        int r = tid & 63;          // edge row within block
        int part = tid >> 6;       // 2 threads per row split the 25 atom pairs
        int e = blk * 64 + r;      // flat edge row (b*L*K + l*K + k)
        int g = e >> 5;            // flat residue b*L + l
        int b = g / LL;
        int j = eidx[e];
        int gj = b * LL + j;
        const float* qa = atoms5 + (size_t)g * 15;
        const float* na = atoms5 + (size_t)gj * 15;
        float qf[15], nf[15];
#pragma unroll
        for (int i = 0; i < 15; ++i) { qf[i] = qa[i]; nf[i] = na[i]; }
        _Float16* row = eh + r * FPAD;

        if (part == 0) {
            // positional encoding: one-hot @ W_pos == column select
            int d = Ridx[gj] - Ridx[g] + MAXREL;
            d = d < 0 ? 0 : (d > 2 * MAXREL ? 2 * MAXREL : d);
            int dfin = (chain[g] == chain[gj]) ? d : (2 * MAXREL + 1);
#pragma unroll
            for (int pp = 0; pp < NPOSE; pp += 2) {
                float e0 = Wpos[pp * (2 * MAXREL + 2) + dfin] + bpos[pp];
                float e1 = Wpos[(pp + 1) * (2 * MAXREL + 2) + dfin] + bpos[pp + 1];
                v2h pk; pk[0] = (_Float16)e0; pk[1] = (_Float16)e1;
                *(v2h*)&row[pp] = pk;
            }
        }
        int q0 = part ? 12 : 0;
        int q1 = part ? 25 : 12;
        for (int qp = q0; qp < q1; ++qp) {
            int a1 = qp / 5, a2 = qp - a1 * 5;
            float dx = qf[a1 * 3 + 0] - nf[a2 * 3 + 0];
            float dy = qf[a1 * 3 + 1] - nf[a2 * 3 + 1];
            float dz = qf[a1 * 3 + 2] - nf[a2 * 3 + 2];
            float D = sqrtf(dx * dx + dy * dy + dz * dz + 1e-6f);
            int off = NPOSE + qp * NRBF;
#pragma unroll
            for (int m = 0; m < NRBF; m += 2) {
                float t0 = D - (2.0f + (float)m * (4.0f / 3.0f));
                float t1 = D - (2.0f + (float)(m + 1) * (4.0f / 3.0f));
                float e0 = __expf(-t0 * t0 * (1.0f / 1.5625f));
                float e1 = __expf(-t1 * t1 * (1.0f / 1.5625f));
                v2h pk; pk[0] = (_Float16)e0; pk[1] = (_Float16)e1;
                *(v2h*)&row[off + m] = pk;
            }
        }
    }
    __syncthreads();

    // ---------------- Phase 2: WMMA GEMM [64 x 416] x [416 x 128] ----------
    const int lane = tid & 31;
    const int w = tid >> 5;            // wave -> 16-row M tile
    const int col = lane & 15;
    const int koff = (lane & 16) ? 8 : 0;   // ISA f16 A/B lane K packing

    v8f acc[8];
#pragma unroll
    for (int t = 0; t < 8; ++t) acc[t] = (v8f){0, 0, 0, 0, 0, 0, 0, 0};

    const _Float16* erow = eh + (w * 16 + col) * FPAD + koff;
    for (int s = 0; s < FEAT / 32; ++s) {
        v8h a0 = *(const v8h*)(erow + 32 * s);
        v8h a1 = *(const v8h*)(erow + 32 * s + 16);
        v16h A = __builtin_shufflevector(a0, a1, 0, 1, 2, 3, 4, 5, 6, 7,
                                         8, 9, 10, 11, 12, 13, 14, 15);
#pragma unroll
        for (int t = 0; t < 8; ++t) {
            // B column n = W_edge row (16t+col), contiguous over features
            const _Float16* wb = wedge + (size_t)(t * 16 + col) * FEAT + 32 * s + koff;
            v8h b0 = *(const v8h*)wb;
            v8h b1 = *(const v8h*)(wb + 16);
            v16h Bm = __builtin_shufflevector(b0, b1, 0, 1, 2, 3, 4, 5, 6, 7,
                                              8, 9, 10, 11, 12, 13, 14, 15);
            acc[t] = __builtin_amdgcn_wmma_f32_16x16x32_f16(
                false, A, false, Bm, (short)0, acc[t], false, false);
        }
    }

    // ---------------- Phase 3: fused LayerNorm + store ---------------------
    float gm[8], bt[8];
#pragma unroll
    for (int t = 0; t < 8; ++t) {
        gm[t] = gam[t * 16 + col];
        bt[t] = bet[t * 16 + col];
    }
    // C/D layout: VGPR m holds row m (lanes 0-15) / row m+8 (lanes 16-31)
    int rbase = blk * 64 + w * 16 + ((lane & 16) ? 8 : 0);
#pragma unroll
    for (int m = 0; m < 8; ++m) {
        float s = 0.f, ss = 0.f;
#pragma unroll
        for (int t = 0; t < 8; ++t) { float v = acc[t][m]; s += v; ss += v * v; }
#pragma unroll
        for (int msk = 8; msk >= 1; msk >>= 1) {   // reduce within 16-lane half
            s  += __shfl_xor(s, msk, 32);
            ss += __shfl_xor(ss, msk, 32);
        }
        float mean = s * (1.0f / 128.0f);
        float var = ss * (1.0f / 128.0f) - mean * mean;
        float inv = rsqrtf(var + 1e-5f);
        float* out = Eout + (size_t)(rbase + m) * HIDN + col;
#pragma unroll
        for (int t = 0; t < 8; ++t)
            out[t * 16] = (acc[t][m] - mean) * inv * gm[t] + bt[t];
    }
}

// ---------------------------------------------------------------------------
extern "C" void kernel_launch(void* const* d_in, const int* in_sizes, int n_in,
                              void* d_out, int out_size, void* d_ws, size_t ws_size,
                              hipStream_t stream) {
    const float* X     = (const float*)d_in[0];
    const float* mask  = (const float*)d_in[1];
    const int*   Ridx  = (const int*)d_in[2];
    const int*   chain = (const int*)d_in[3];
    const float* Wpos  = (const float*)d_in[4];
    const float* bpos  = (const float*)d_in[5];
    const float* Wedge = (const float*)d_in[6];
    const float* gam   = (const float*)d_in[7];
    const float* bet   = (const float*)d_in[8];

    int*   eidx = (int*)d_out;                                 // E_idx (int32 bits)
    float* Eout = (float*)d_out + (size_t)BB * LL * KK;        // E

    float*     atoms5  = (float*)d_ws;                               // B*L*15 f32
    _Float16*  wedge16 = (_Float16*)((char*)d_ws +
                                     (size_t)BB * LL * 15 * sizeof(float));

    geom_kernel<<<(BB * LL + 255) / 256, 256, 0, stream>>>(X, atoms5, BB * LL);
    cvt_kernel<<<(HIDN * FEAT + 255) / 256, 256, 0, stream>>>(Wedge, wedge16,
                                                              HIDN * FEAT);
    topk_kernel<<<(BB * LL) / 4, 128, 0, stream>>>(atoms5, mask, eidx);
    edge_kernel<<<(BB * LL * KK) / 64, 128, 0, stream>>>(
        atoms5, wedge16, Ridx, chain, Wpos, bpos, gam, bet, eidx, Eout);
}
